// CPCModel_12644383719492
// MI455X (gfx1250) — matrix operand order repaired and checked by reading, not compiled, not run on hardware
//
#include <hip/hip_runtime.h>
#include <hip/hip_bf16.h>
#include <math.h>

typedef __bf16 bf16_t;
typedef __attribute__((ext_vector_type(16))) __bf16 v16bf;
typedef __attribute__((ext_vector_type(8)))  float  v8f;
typedef __attribute__((ext_vector_type(4)))  unsigned int u32x4;
typedef __attribute__((ext_vector_type(8)))  int          i32x8;
typedef __attribute__((ext_vector_type(4)))  int          i32x4;

static __device__ __forceinline__ bf16_t to_bf16(float f) { return (bf16_t)f; }

// ---------------------------------------------------------------------------
// Layer 1 conv: C_in=12, K=4, stride 2, pad 1.  (8,12,8192) -> (8,64,4096)
// Small reduction (48 MACs) -> plain VALU kernel. Bias omitted (BN-invariant).
// ---------------------------------------------------------------------------
__global__ void conv1_kernel(const float* __restrict__ x,
                             const float* __restrict__ w,
                             float* __restrict__ out) {
  const int L_in = 8192, L_out = 4096, CIN = 12, COUT = 64;
  int idx = blockIdx.x * blockDim.x + threadIdx.x;
  int total = 8 * COUT * L_out;
  if (idx >= total) return;
  int l  = idx % L_out;
  int co = (idx / L_out) % COUT;
  int b  = idx / (L_out * COUT);
  float acc = 0.f;
  const float* wr = w + co * CIN * 4;
  const float* xb = x + b * CIN * L_in;
  #pragma unroll
  for (int k = 0; k < 4; ++k) {
    int p = 2 * l - 1 + k;
    if (p < 0 || p >= L_in) continue;
    #pragma unroll 4
    for (int ci = 0; ci < CIN; ++ci)
      acc = fmaf(wr[ci * 4 + k], xb[ci * L_in + p], acc);
  }
  out[idx] = acc;
}

// ---------------------------------------------------------------------------
// Training-mode BN statistics: one block per channel, reduce over B*L.
// stats[2c] = mean, stats[2c+1] = 1/sqrt(var+eps)
// ---------------------------------------------------------------------------
__global__ void stats_kernel(const float* __restrict__ X,
                             float* __restrict__ stats,
                             int C, int L, int B) {
  __shared__ float ssum[256];
  __shared__ float ssq[256];
  int c = blockIdx.x;
  int n = B * L;
  float s = 0.f, s2 = 0.f;
  for (int i = threadIdx.x; i < n; i += blockDim.x) {
    int b = i / L, l = i - b * L;
    float v = X[(b * C + c) * L + l];
    s += v; s2 += v * v;
  }
  ssum[threadIdx.x] = s; ssq[threadIdx.x] = s2;
  __syncthreads();
  for (int o = 128; o > 0; o >>= 1) {
    if ((int)threadIdx.x < o) {
      ssum[threadIdx.x] += ssum[threadIdx.x + o];
      ssq[threadIdx.x]  += ssq[threadIdx.x + o];
    }
    __syncthreads();
  }
  if (threadIdx.x == 0) {
    float inv_n = 1.f / (float)n;
    float m = ssum[0] * inv_n;
    float v = ssq[0] * inv_n - m * m;
    stats[2 * c]     = m;
    stats[2 * c + 1] = rsqrtf(v + 1e-5f);
  }
}

// Normalize + ReLU, emit bf16 operand for next WMMA conv layer.
__global__ void bnrelu_bf16_kernel(const float* __restrict__ X,
                                   const float* __restrict__ stats,
                                   const float* __restrict__ g,
                                   const float* __restrict__ be,
                                   bf16_t* __restrict__ out,
                                   int C, int L, int total) {
  int i = blockIdx.x * blockDim.x + threadIdx.x;
  if (i >= total) return;
  int c = (i / L) % C;
  float v = (X[i] - stats[2 * c]) * stats[2 * c + 1] * g[c] + be[c];
  out[i] = to_bf16(fmaxf(v, 0.f));
}

// Layer-3 normalize + ReLU, write transposed z=(B,L,D) f32 straight into d_out.
__global__ void bnrelu_z_kernel(const float* __restrict__ X,
                                const float* __restrict__ stats,
                                const float* __restrict__ g,
                                const float* __restrict__ be,
                                float* __restrict__ z) {
  const int C = 128, L = 1024;
  int i = blockIdx.x * blockDim.x + threadIdx.x;
  if (i >= 8 * C * L) return;
  int l = i % L;
  int c = (i / L) % C;
  int b = i / (C * L);
  float v = (X[i] - stats[2 * c]) * stats[2 * c + 1] * g[c] + be[c];
  z[(b * L + l) * C + c] = fmaxf(v, 0.f);
}

__global__ void f32_to_bf16_kernel(const float* __restrict__ in,
                                   bf16_t* __restrict__ out, int n) {
  int i = blockIdx.x * blockDim.x + threadIdx.x;
  if (i < n) out[i] = to_bf16(in[i]);
}

// ---------------------------------------------------------------------------
// Conv layers 2/3 as implicit-im2col GEMM on v_wmma_f32_16x16x32_bf16.
//   Y[co, col] = sum_r W[co, r] * X[b, r>>2, 2l-1+(r&3)],  r = ci*4+k.
// Block = 256 threads = 8 waves; block tile = 64(M) x 32(N); KRED = 256.
//
// Data movement (MI455X path):
//   * Weight tile (64 x 256 bf16 = 32KB) DMA'd into LDS by the Tensor Data
//     Mover (tensor_load_to_lds, D# built per ISA §8), with LDS row padding
//     (pad_interval = 128 dwords, pad_amount = 4 dwords -> row stride 264
//     elements) so A-fragment ds reads spread across banks. Drained with
//     s_wait_tensorcnt 0 before the block barrier.
//   * Activation window (CIN x 68 positions, ~9KB) staged cooperatively with
//     coalesced global loads; B fragments then come from LDS.
// Fragment k order: contiguous, k = 16*half + s, same for A and B.
// ---------------------------------------------------------------------------
template <int CIN, int COUT, int LIN, int LOUT>
__global__ void conv_wmma_kernel(const bf16_t* __restrict__ X,
                                 const bf16_t* __restrict__ W,
                                 float* __restrict__ Y) {
  constexpr int KRED  = CIN * 4;     // 256
  constexpr int NK    = KRED / 32;   // 8 WMMA K-steps
  constexpr int KREDP = KRED + 8;    // padded LDS row stride (16B-aligned rows)
  constexpr int QN    = 68;          // input positions needed for 32 columns
  constexpr int PQ    = 72;          // padded q stride

  __shared__ bf16_t wlds[64 * KREDP];
  __shared__ bf16_t xs[CIN * PQ];

  int tid  = threadIdx.x;
  int wave = tid >> 5;
  int lane = tid & 31;

  int m_blk = blockIdx.y * 64;
  int c0 = blockIdx.x * 32;          // first output column of this block
  int b  = c0 / LOUT;
  int l0 = c0 - b * LOUT;
  int pbase = 2 * l0 - 1;

  // ---- TDM: DMA weight tile rows [m_blk, m_blk+64) into LDS (wave 0) ----
#if __has_builtin(__builtin_amdgcn_tensor_load_to_lds)
  if (wave == 0) {
    unsigned long long gaddr = (unsigned long long)(const void*)(W + (size_t)m_blk * KRED);
    unsigned int ldsoff = (unsigned int)(size_t)(void*)wlds;
    u32x4 g0;
    g0[0] = 1u;                                        // count=1, user D#
    g0[1] = ldsoff;                                    // lds_addr
    g0[2] = (unsigned int)(gaddr & 0xFFFFFFFFu);       // global_addr lo
    g0[3] = (unsigned int)((gaddr >> 32) & 0x1FFFFFFu) // global_addr hi
            | (2u << 30);                              // type = 2 ("image")
    i32x8 g1;
    const unsigned int td0 = KRED, td1 = 64;           // tensor dims
    const unsigned int tl0 = KRED, tl1 = 64;           // tile dims
    const unsigned long long s0 = KRED;                // tensor_dim0_stride
    g1[0] = (int)((1u << 16)      // data_size = 2B
                | (1u << 20)      // pad_enable
                | (6u << 22)      // pad_interval: 2^(6+1)=128 dwords (=512B row)
                | (3u << 25));    // pad_amount: 4 dwords (=8 bf16 elements)
    g1[1] = (int)((td0 & 0xFFFFu) << 16);              // ab_addr=0 | td0.lo16
    g1[2] = (int)((td0 >> 16) | ((td1 & 0xFFFFu) << 16));
    g1[3] = (int)((td1 >> 16) | (tl0 << 16));          // td1.hi | tile_dim0
    g1[4] = (int)(tl1);                                // tile_dim1 | tile_dim2=0
    g1[5] = (int)(s0 & 0xFFFFFFFFu);
    g1[6] = (int)((s0 >> 32) & 0xFFFFu);               // s0.hi16 | dim1_stride.lo=0
    g1[7] = 0;
    i32x4 gz = {0, 0, 0, 0};
#if defined(__clang_major__) && __clang_major__ >= 23
    i32x8 gz8 = {0, 0, 0, 0, 0, 0, 0, 0};
    __builtin_amdgcn_tensor_load_to_lds(g0, g1, gz, gz, gz8, 0);
#else
    __builtin_amdgcn_tensor_load_to_lds(g0, g1, gz, gz, 0);
#endif
  }
#else
  for (int i = tid; i < 64 * KRED; i += 256) {
    int row = i / KRED, col = i - row * KRED;
    wlds[row * KREDP + col] = W[(size_t)(m_blk + row) * KRED + col];
  }
#endif

  // ---- Stage activation window into LDS (coalesced, zero-filled halo) ----
  for (int e = tid; e < CIN * QN; e += 256) {
    int ci = e / QN, q = e - ci * QN;
    int p = pbase + q;
    bf16_t v = (bf16_t)0.f;
    if (p >= 0 && p < LIN) v = X[((size_t)b * CIN + ci) * LIN + p];
    xs[ci * PQ + q] = v;
  }

#if __has_builtin(__builtin_amdgcn_s_wait_tensorcnt)
  __builtin_amdgcn_s_wait_tensorcnt(0);   // no-op for non-issuing waves
#endif
  __syncthreads();

  // ---- WMMA main loop ----
  int mt = wave & 3;
  int nt = wave >> 2;
  int half = lane >> 4;
  int lrow = lane & 15;
  int dl = nt * 16 + lrow;               // column within block, 0..31

  const bf16_t* wrow = wlds + (mt * 16 + lrow) * KREDP;

  v8f acc = {};
  #pragma unroll
  for (int kk = 0; kk < NK; ++kk) {
    v16bf afrag, bfrag;
    int kb = kk * 32 + 16 * half;
    #pragma unroll
    for (int e = 0; e < 8; ++e) {        // two contiguous 8-element LDS runs
      afrag[e]     = wrow[kb + e];
      afrag[e + 8] = wrow[kb + 8 + e];
    }
    int ciBase = kk * 8 + 4 * half;      // (kk*32 + 16*half) >> 2
    #pragma unroll
    for (int j = 0; j < 4; ++j) {
      int base = (ciBase + j) * PQ + 2 * dl;
      #pragma unroll
      for (int k = 0; k < 4; ++k)
        bfrag[4 * j + k] = xs[base + k];
    }
    acc = __builtin_amdgcn_wmma_f32_16x16x32_bf16(
        /*neg_a=*/false, afrag, /*neg_b=*/false, bfrag,
        /*c_mod=*/(short)0, acc, /*reuse_a=*/false, /*reuse_b=*/false);
  }

  // ---- Store C tile (bias omitted: BN-invariant) ----
  int nout = c0 + nt * 16 + lrow;
  int bo = nout / LOUT;
  int lo = nout - bo * LOUT;
  #pragma unroll
  for (int r = 0; r < 8; ++r) {
    int m = m_blk + mt * 16 + 8 * half + r;
    Y[((size_t)bo * COUT + m) * LOUT + lo] = acc[r];
  }
}

// ---------------------------------------------------------------------------
// S4 diagonal complex SSM scan. One wave32 per (b,d) pair (1024 waves),
// 2 complex states per lane (N=64). y_t = Re(sum_n C h) via shfl_xor butterfly.
//   A_d = 1 + (-exp(log_A_real) + i*A_imag)*dt,  B_d = B*dt (real input u).
// ---------------------------------------------------------------------------
__global__ void s4_scan_kernel(const float* __restrict__ z,
                               const float* __restrict__ log_A_real,
                               const float* __restrict__ A_imag,
                               const float* __restrict__ Bmat,
                               const float* __restrict__ Cmat,
                               const float* __restrict__ Dvec,
                               const float* __restrict__ log_dt,
                               float* __restrict__ y) {
  const int D = 128, N = 64, L = 1024;
  int wave = threadIdx.x >> 5;
  int lane = threadIdx.x & 31;
  int wg = blockIdx.x * 8 + wave;   // 0..1023
  int b = wg >> 7;
  int d = wg & 127;
  float dt = __expf(log_dt[0]);

  float adr[2], adi[2], bd[2], cr[2], ci[2];
  float hr[2] = {0.f, 0.f}, hi[2] = {0.f, 0.f};
  #pragma unroll
  for (int j = 0; j < 2; ++j) {
    int idx = d * N + lane + 32 * j;
    adr[j] = 1.f - __expf(log_A_real[idx]) * dt;
    adi[j] = A_imag[idx] * dt;
    bd[j]  = Bmat[idx] * dt;
    cr[j]  = Cmat[2 * idx];
    ci[j]  = Cmat[2 * idx + 1];
  }
  float Dd = Dvec[d];
  const float* zu = z + (size_t)(b * L) * D + d;
  float* yo = y + (size_t)(b * L) * D + d;

  for (int l = 0; l < L; ++l) {
    if ((l & 63) == 0 && l + 64 < L)
      __builtin_prefetch(zu + (size_t)(l + 64) * D, 0, 0);
    float u = zu[(size_t)l * D];
    float p = 0.f;
    #pragma unroll
    for (int j = 0; j < 2; ++j) {
      float nr = adr[j] * hr[j] - adi[j] * hi[j] + bd[j] * u;
      float ni = adr[j] * hi[j] + adi[j] * hr[j];
      hr[j] = nr; hi[j] = ni;
      p += nr * cr[j] - ni * ci[j];
    }
    #pragma unroll
    for (int off = 16; off > 0; off >>= 1)
      p += __shfl_xor(p, off, 32);
    if (lane == 0)
      yo[(size_t)l * D] = p + Dd * u;
  }
}

// LayerNorm over D=128: one wave per (b,l) row, 4 values per lane.
__global__ void layernorm_kernel(const float* __restrict__ y,
                                 const float* __restrict__ g,
                                 const float* __restrict__ be,
                                 float* __restrict__ out) {
  const int D = 128;
  int wave = threadIdx.x >> 5;
  int lane = threadIdx.x & 31;
  int row = blockIdx.x * 8 + wave;   // 8192 rows
  const float* yr = y + (size_t)row * D;
  float v[4];
  float s = 0.f, s2 = 0.f;
  #pragma unroll
  for (int j = 0; j < 4; ++j) {
    v[j] = yr[lane + 32 * j];
    s += v[j]; s2 += v[j] * v[j];
  }
  #pragma unroll
  for (int off = 16; off > 0; off >>= 1) {
    s  += __shfl_xor(s, off, 32);
    s2 += __shfl_xor(s2, off, 32);
  }
  float mu  = s * (1.f / D);
  float var = s2 * (1.f / D) - mu * mu;
  float inv = rsqrtf(var + 1e-5f);
  float* outr = out + (size_t)row * D;
  #pragma unroll
  for (int j = 0; j < 4; ++j) {
    int dd = lane + 32 * j;
    outr[dd] = (v[j] - mu) * inv * g[dd] + be[dd];
  }
}

// ---------------------------------------------------------------------------
extern "C" void kernel_launch(void* const* d_in, const int* in_sizes, int n_in,
                              void* d_out, int out_size, void* d_ws, size_t ws_size,
                              hipStream_t stream) {
  (void)in_sizes; (void)n_in; (void)out_size; (void)ws_size;
  const float* x          = (const float*)d_in[0];
  const float* w1         = (const float*)d_in[1];
  const float* g1         = (const float*)d_in[3];
  const float* be1        = (const float*)d_in[4];
  const float* w2         = (const float*)d_in[5];
  const float* g2         = (const float*)d_in[7];
  const float* be2        = (const float*)d_in[8];
  const float* w3         = (const float*)d_in[9];
  const float* g3         = (const float*)d_in[11];
  const float* be3        = (const float*)d_in[12];
  const float* log_A_real = (const float*)d_in[13];
  const float* A_imag     = (const float*)d_in[14];
  const float* Bmat       = (const float*)d_in[15];
  const float* Cmat       = (const float*)d_in[16];
  const float* Dvec       = (const float*)d_in[17];
  const float* log_dt     = (const float*)d_in[18];
  const float* ln_g       = (const float*)d_in[19];
  const float* ln_b       = (const float*)d_in[20];

  char* ws = (char*)d_ws;
  size_t off = 0;
  auto alloc = [&](size_t bytes) -> char* {
    char* p = ws + off;
    off += (bytes + 255) & ~(size_t)255;
    return p;
  };
  float*  c1  = (float*) alloc((size_t)8 * 64 * 4096 * 4);
  bf16_t* h1b = (bf16_t*)alloc((size_t)8 * 64 * 4096 * 2);
  float*  c2  = (float*) alloc((size_t)8 * 64 * 2048 * 4);
  bf16_t* h2b = (bf16_t*)alloc((size_t)8 * 64 * 2048 * 2);
  float*  c3  = (float*) alloc((size_t)8 * 128 * 1024 * 4);
  float*  yb  = (float*) alloc((size_t)8 * 1024 * 128 * 4);
  bf16_t* w2b = (bf16_t*)alloc((size_t)64 * 256 * 2);
  bf16_t* w3b = (bf16_t*)alloc((size_t)128 * 256 * 2);
  float*  st1 = (float*) alloc(2 * 64 * 4);
  float*  st2 = (float*) alloc(2 * 64 * 4);
  float*  st3 = (float*) alloc(2 * 128 * 4);

  float* zout = (float*)d_out;            // z: (8,1024,128)
  float* cout = zout + (size_t)8 * 1024 * 128;

  // Weight precision conversion for WMMA layers
  f32_to_bf16_kernel<<<(16384 + 255) / 256, 256, 0, stream>>>(w2, w2b, 16384);
  f32_to_bf16_kernel<<<(32768 + 255) / 256, 256, 0, stream>>>(w3, w3b, 32768);

  // Layer 1
  conv1_kernel<<<(8 * 64 * 4096) / 256, 256, 0, stream>>>(x, w1, c1);
  stats_kernel<<<64, 256, 0, stream>>>(c1, st1, 64, 4096, 8);
  bnrelu_bf16_kernel<<<(8 * 64 * 4096) / 256, 256, 0, stream>>>(
      c1, st1, g1, be1, h1b, 64, 4096, 8 * 64 * 4096);

  // Layer 2 (WMMA GEMM: M=64, K=256, N=16384)
  conv_wmma_kernel<64, 64, 4096, 2048>
      <<<dim3(16384 / 32, 1), 256, 0, stream>>>(h1b, w2b, c2);
  stats_kernel<<<64, 256, 0, stream>>>(c2, st2, 64, 2048, 8);
  bnrelu_bf16_kernel<<<(8 * 64 * 2048) / 256, 256, 0, stream>>>(
      c2, st2, g2, be2, h2b, 64, 2048, 8 * 64 * 2048);

  // Layer 3 (WMMA GEMM: M=128, K=256, N=8192)
  conv_wmma_kernel<64, 128, 2048, 1024>
      <<<dim3(8192 / 32, 2), 256, 0, stream>>>(h2b, w3b, c3);
  stats_kernel<<<128, 256, 0, stream>>>(c3, st3, 128, 1024, 8);
  bnrelu_z_kernel<<<(8 * 128 * 1024) / 256, 256, 0, stream>>>(
      c3, st3, g3, be3, zout);

  // S4 scan + LayerNorm
  s4_scan_kernel<<<128, 256, 0, stream>>>(zout, log_A_real, A_imag, Bmat,
                                          Cmat, Dvec, log_dt, yb);
  layernorm_kernel<<<8192 / 8, 256, 0, stream>>>(yb, ln_g, ln_b, cout);
}
